// RGCN_84482006712681
// MI455X (gfx1250) — matrix operand order repaired and checked by reading, not compiled, not run on hardware
//
#include <hip/hip_runtime.h>
#include <hip/hip_bf16.h>

// ---------------- constants from the reference ----------------
enum { N_ = 20000, E_ = 320000, B_ = 64, DT_ = 768, DP_ = 128, H_ = 256, O_ = 128, NREL = 2 };

typedef __attribute__((ext_vector_type(16))) __bf16 v16bf;
typedef __attribute__((ext_vector_type(8)))  __bf16 v8bf;
typedef __attribute__((ext_vector_type(8)))  float  v8f;

__device__ __forceinline__ unsigned short f2bf(float f) {
    unsigned u = __float_as_uint(f);
    unsigned r = u + 0x7FFFu + ((u >> 16) & 1u);   // RNE
    return (unsigned short)(r >> 16);
}

// ---------------- f32 -> bf16 conversion ----------------
__global__ void f32_to_bf16_kernel(const float* __restrict__ x,
                                   unsigned short* __restrict__ y, size_t n) {
    size_t i = (size_t)blockIdx.x * blockDim.x + threadIdx.x;
    if (i < n) y[i] = f2bf(x[i]);
}

// ---------------- weight repack for WMMA B-fragments ----------------
// Packed layout: [kt][nt][lane][i], 512 bf16 per 32x16 K-N tile.
// Lane L supplies column n = nt*16 + (L%16), K = kt*32 + 8*(L/16) + i (i<8), kt*32 + 16 + 8*(L/16) + (i-8) (i>=8).
__global__ void pack_b_kernel(const float* __restrict__ w,
                              unsigned short* __restrict__ wp, int din, int dout) {
    size_t idx = (size_t)blockIdx.x * blockDim.x + threadIdx.x;
    size_t total = (size_t)din * dout;
    if (idx >= total) return;
    int i    = (int)(idx & 15);
    int lane = (int)((idx >> 4) & 31);
    size_t t = idx >> 9;                   // tile index = kt*NT + nt
    int NT = dout >> 4;
    int nt = (int)(t % NT), kt = (int)(t / NT);
    int n = nt * 16 + (lane & 15);
    int k = kt * 32 + ((lane >> 4) << 3) + (i < 8 ? i : 8 + i);
    wp[idx] = f2bf(w[(size_t)k * dout + n]);
}

// ---------------- bf16 WMMA GEMM: C[M,dout] = A[M,din] * B (+bias) ----------------
// 256 threads = 8 waves (4 M x 2 N); wave tile 32x64; block tile 128x128.
// All WMMAs execute unconditionally (EXEC all-1s); the M-tail is handled by
// clamping A-load rows to M-1 and guarding only the stores (wave-uniform).
__global__ __launch_bounds__(256)
void gemm_bf16_kernel(const __bf16* __restrict__ A, const __bf16* __restrict__ Bp,
                      const float* __restrict__ bias, float* __restrict__ C,
                      int M, int din, int dout) {
    const int lane = threadIdx.x & 31;
    const int wave = threadIdx.x >> 5;
    const int wm = wave >> 1, wn = wave & 1;
    const int nbn = dout >> 7;                       // 128-wide N block tiles
    const int bm = blockIdx.x / nbn, bn = blockIdx.x % nbn;
    const int m0 = bm * 128 + wm * 32;
    if (m0 >= M) return;                             // wave-uniform early out
    const bool mv1 = (m0 + 16) < M;                  // wave-uniform
    const int n0 = bn * 128 + wn * 64;               // wave covers 4 N-tiles
    const int NT = dout >> 4;
    const int nt0 = n0 >> 4;
    const int lrow  = lane & 15;
    const int khalf = (lane >> 4) << 3;              // 0 or 8

    int row0 = m0 + lrow;
    int row1 = m0 + 16 + lrow;
    if (row1 >= M) row1 = M - 1;                     // clamp: garbage computed, never stored
    const size_t arow0 = (size_t)row0 * din;
    const size_t arow1 = (size_t)row1 * din;

    v8f acc0[4] = {{}, {}, {}, {}};
    v8f acc1[4] = {{}, {}, {}, {}};

    const int KT = din >> 5;
    for (int kt = 0; kt < KT; ++kt) {
        const int kb = (kt << 5) + khalf;
        v16bf a0, a1;
        {   // A frags: ISA 16-bit A layout. lo = K[khalf..+7], hi = K[16+khalf..]
            const v8bf* pa0 = (const v8bf*)(A + arow0 + kb);
            const v8bf* pa1 = (const v8bf*)(A + arow1 + kb);
            v8bf l0 = pa0[0], h0 = pa0[2];
            v8bf l1 = pa1[0], h1 = pa1[2];
            #pragma unroll
            for (int i = 0; i < 8; ++i) {
                a0[i] = l0[i]; a0[8 + i] = h0[i];
                a1[i] = l1[i]; a1[8 + i] = h1[i];
            }
        }
        if (kt + 1 < KT) {                            // gfx1250 global_prefetch_b8
            __builtin_prefetch(A + arow0 + kb + 32, 0, 0);
            __builtin_prefetch(A + arow1 + kb + 32, 0, 0);
        }
        v16bf b[4];
        #pragma unroll
        for (int j = 0; j < 4; ++j) {                 // B frags: contiguous per-lane
            const v8bf* pb = (const v8bf*)(Bp + (((size_t)kt * NT + nt0 + j) * 32 + lane) * 16);
            v8bf lo = pb[0], hi = pb[1];
            #pragma unroll
            for (int i = 0; i < 8; ++i) { b[j][i] = lo[i]; b[j][8 + i] = hi[i]; }
        }
        #pragma unroll
        for (int j = 0; j < 4; ++j)
            acc0[j] = __builtin_amdgcn_wmma_f32_16x16x32_bf16(false, a0, false, b[j], (short)0, acc0[j], false, false);
        #pragma unroll
        for (int j = 0; j < 4; ++j)
            acc1[j] = __builtin_amdgcn_wmma_f32_16x16x32_bf16(false, a1, false, b[j], (short)0, acc1[j], false, false);
    }

    // C/D layout: elem v of lane -> row m0 + khalf + v, col = ntile*16 + (lane&15)
    #pragma unroll
    for (int j = 0; j < 4; ++j) {
        const int c = n0 + j * 16 + lrow;
        const float bi = bias ? bias[c] : 0.0f;
        #pragma unroll
        for (int v = 0; v < 8; ++v) {
            size_t r = (size_t)(m0 + khalf + v) * dout;
            C[r + c] = acc0[j][v] + bi;
        }
    }
    if (mv1) {
        #pragma unroll
        for (int j = 0; j < 4; ++j) {
            const int c = n0 + j * 16 + lrow;
            const float bi = bias ? bias[c] : 0.0f;
            #pragma unroll
            for (int v = 0; v < 8; ++v) {
                size_t r = (size_t)(m0 + 16 + khalf + v) * dout;
                C[r + c] = acc1[j][v] + bi;
            }
        }
    }
}

// ---------------- edge scatter (per-relation mean numerator + counts) ----------------
__global__ void edge_agg_kernel(const int* __restrict__ ei, const int* __restrict__ et,
                                const float* __restrict__ h, float* __restrict__ acc,
                                float* __restrict__ cnt, int E, int dout, int rel) {
    int e = (int)(((size_t)blockIdx.x * blockDim.x + threadIdx.x) >> 5);
    int lane = threadIdx.x & 31;
    if (e >= E) return;
    if (et[e] != rel) return;
    int src = ei[e];
    int dst = ei[E + e];
    const float* hs = h + (size_t)src * dout;
    float* ad = acc + (size_t)dst * dout;
    for (int j = lane; j < dout; j += 32) atomicAdd(&ad[j], hs[j]);
    if (lane == 0) atomicAdd(&cnt[dst], 1.0f);
}

__global__ void combine_kernel(float* __restrict__ out, const float* __restrict__ acc,
                               const float* __restrict__ cnt, int N, int dout) {
    size_t i = (size_t)blockIdx.x * blockDim.x + threadIdx.x;
    if (i >= (size_t)N * dout) return;
    int row = (int)(i / dout);
    float c = cnt[row]; c = c < 1.0f ? 1.0f : c;
    out[i] += acc[i] / c;
}

// ---------------- batchnorm (two pass) + leaky relu + bf16 emit ----------------
__global__ void colsum_kernel(const float* __restrict__ x, float* __restrict__ stats,
                              int N, int dout) {
    int rpb = blockDim.x / dout;               // rows per iteration (dout divides 256)
    int col = threadIdx.x % dout;
    int r0  = threadIdx.x / dout;
    float s = 0.0f, ss = 0.0f;
    for (int row = blockIdx.x * rpb + r0; row < N; row += gridDim.x * rpb) {
        float v = x[(size_t)row * dout + col];
        s += v; ss += v * v;
    }
    atomicAdd(&stats[col], s);
    atomicAdd(&stats[dout + col], ss);
}

__global__ void bn_apply_kernel(float* __restrict__ x, const float* __restrict__ stats,
                                const float* __restrict__ gamma, const float* __restrict__ beta,
                                unsigned short* __restrict__ xb, int N, int dout) {
    size_t i = (size_t)blockIdx.x * blockDim.x + threadIdx.x;
    if (i >= (size_t)N * dout) return;
    int col = (int)(i % dout);
    float invN = 1.0f / (float)N;
    float mu  = stats[col] * invN;
    float var = stats[dout + col] * invN - mu * mu;
    float v = (x[i] - mu) * rsqrtf(var + 1e-5f) * gamma[col] + beta[col];
    v = (v >= 0.0f) ? v : 0.01f * v;
    x[i] = v;
    if (xb) xb[i] = f2bf(v);
}

// ---------------- scatter-mean pooling ----------------
__global__ void pool_acc_kernel(const float* __restrict__ h, const int* __restrict__ batch,
                                float* __restrict__ pacc, float* __restrict__ pcnt,
                                int N, int O) {
    int node = (int)(((size_t)blockIdx.x * blockDim.x + threadIdx.x) >> 5);
    int lane = threadIdx.x & 31;
    if (node >= N) return;
    int b = batch[node];
    const float* hs = h + (size_t)node * O;
    float* pd = pacc + (size_t)b * O;
    for (int j = lane; j < O; j += 32) atomicAdd(&pd[j], hs[j]);
    if (lane == 0) atomicAdd(&pcnt[b], 1.0f);
}

__global__ void pool_fin_kernel(const float* __restrict__ pacc, const float* __restrict__ pcnt,
                                float* __restrict__ out, int B, int O) {
    int i = blockIdx.x * blockDim.x + threadIdx.x;
    if (i >= B * O) return;
    float c = pcnt[i / O]; c = c < 1.0f ? 1.0f : c;
    out[i] = pacc[i] / c;
}

// =======================================================================
//                             host orchestration
// =======================================================================
struct WS {
    unsigned short* xb;     // N * 768 bf16 (layer input; reused for next layer's bf16)
    unsigned short* wpack;  // 3 * 768*256 bf16 packed weights
    float* h;               // N * H   (conv output / BN in place)
    float* hb;              // N * H   (per-relation dense GEMM result)
    float* acc;             // N * H   (edge aggregation)
    float* cnt;             // N
    float* stats;           // 2 * H
    float* pacc;            // B * O
    float* pcnt;            // B
};

static inline void launch_gemm(const unsigned short* A, const unsigned short* Bp,
                               const float* bias, float* C,
                               int M, int din, int dout, hipStream_t s) {
    int grid = ((M + 127) / 128) * (dout / 128);
    gemm_bf16_kernel<<<grid, 256, 0, s>>>((const __bf16*)A, (const __bf16*)Bp, bias, C, M, din, dout);
}

static void run_conv(const unsigned short* xb, int din, int dout,
                     const float* w_rel, const float* w_root, const float* bias,
                     const int* ei, const int* et,
                     WS& ws, hipStream_t stream) {
    size_t wsz = (size_t)din * dout;
    int pk_blocks = (int)((wsz + 255) / 256);
    pack_b_kernel<<<pk_blocks, 256, 0, stream>>>(w_root,        ws.wpack,           din, dout);
    pack_b_kernel<<<pk_blocks, 256, 0, stream>>>(w_rel,         ws.wpack + wsz,     din, dout);
    pack_b_kernel<<<pk_blocks, 256, 0, stream>>>(w_rel + wsz,   ws.wpack + 2 * wsz, din, dout);

    launch_gemm(xb, ws.wpack, bias, ws.h, N_, din, dout, stream);     // root + bias

    int eblocks = (E_ * 32 + 255) / 256;
    int cblocks = (int)(((size_t)N_ * dout + 255) / 256);
    for (int r = 0; r < NREL; ++r) {
        launch_gemm(xb, ws.wpack + (size_t)(1 + r) * wsz, nullptr, ws.hb, N_, din, dout, stream);
        hipMemsetAsync(ws.acc, 0, (size_t)N_ * dout * sizeof(float), stream);
        hipMemsetAsync(ws.cnt, 0, (size_t)N_ * sizeof(float), stream);
        edge_agg_kernel<<<eblocks, 256, 0, stream>>>(ei, et, ws.hb, ws.acc, ws.cnt, E_, dout, r);
        combine_kernel<<<cblocks, 256, 0, stream>>>(ws.h, ws.acc, ws.cnt, N_, dout);
    }
}

static void run_bn(float* h, const float* gamma, const float* beta,
                   unsigned short* xb_out, WS& ws, int dout, hipStream_t stream) {
    hipMemsetAsync(ws.stats, 0, 2 * dout * sizeof(float), stream);
    colsum_kernel<<<240, 256, 0, stream>>>(h, ws.stats, N_, dout);
    int blocks = (int)(((size_t)N_ * dout + 255) / 256);
    bn_apply_kernel<<<blocks, 256, 0, stream>>>(h, ws.stats, gamma, beta, xb_out, N_, dout);
}

static void run_branch(const float* x, int din,
                       const int* ei, const int* et, const int* batch,
                       const float* w1_rel, const float* w1_root, const float* b1,
                       const float* w2_rel, const float* w2_root, const float* b2,
                       const float* g1, const float* be1,
                       const float* g2, const float* be2,
                       float* out_pool, WS& ws, hipStream_t stream) {
    size_t nx = (size_t)N_ * din;
    f32_to_bf16_kernel<<<(int)((nx + 255) / 256), 256, 0, stream>>>(x, ws.xb, nx);

    run_conv(ws.xb, din, H_, w1_rel, w1_root, b1, ei, et, ws, stream);
    run_bn(ws.h, g1, be1, ws.xb, ws, H_, stream);       // bf16 output reuses xb

    run_conv(ws.xb, H_, O_, w2_rel, w2_root, b2, ei, et, ws, stream);
    run_bn(ws.h, g2, be2, nullptr, ws, O_, stream);

    hipMemsetAsync(ws.pacc, 0, (size_t)B_ * O_ * sizeof(float), stream);
    hipMemsetAsync(ws.pcnt, 0, (size_t)B_ * sizeof(float), stream);
    pool_acc_kernel<<<(N_ * 32 + 255) / 256, 256, 0, stream>>>(ws.h, batch, ws.pacc, ws.pcnt, N_, O_);
    pool_fin_kernel<<<(B_ * O_ + 255) / 256, 256, 0, stream>>>(ws.pacc, ws.pcnt, out_pool, B_, O_);
}

extern "C" void kernel_launch(void* const* d_in, const int* in_sizes, int n_in,
                              void* d_out, int out_size, void* d_ws, size_t ws_size,
                              hipStream_t stream) {
    // inputs (setup_inputs order)
    const float* x_t        = (const float*)d_in[0];
    const float* x_p        = (const float*)d_in[1];
    const int*   ei_t       = (const int*)d_in[2];
    const int*   et_t       = (const int*)d_in[3];
    const int*   batch_t    = (const int*)d_in[4];
    const int*   ei_p       = (const int*)d_in[5];
    const int*   et_p       = (const int*)d_in[6];
    const int*   batch_p    = (const int*)d_in[7];
    const float* w_t1_rel   = (const float*)d_in[8];
    const float* w_t1_root  = (const float*)d_in[9];
    const float* b_t1       = (const float*)d_in[10];
    const float* w_t2_rel   = (const float*)d_in[11];
    const float* w_t2_root  = (const float*)d_in[12];
    const float* b_t2       = (const float*)d_in[13];
    const float* w_p1_rel   = (const float*)d_in[14];
    const float* w_p1_root  = (const float*)d_in[15];
    const float* b_p1       = (const float*)d_in[16];
    const float* w_p2_rel   = (const float*)d_in[17];
    const float* w_p2_root  = (const float*)d_in[18];
    const float* b_p2       = (const float*)d_in[19];
    const float* g_t1  = (const float*)d_in[20];
    const float* be_t1 = (const float*)d_in[21];
    const float* g_t2  = (const float*)d_in[22];
    const float* be_t2 = (const float*)d_in[23];
    const float* g_p1  = (const float*)d_in[24];
    const float* be_p1 = (const float*)d_in[25];
    const float* g_p2  = (const float*)d_in[26];
    const float* be_p2 = (const float*)d_in[27];

    // workspace carving (256-byte aligned)
    char* p = (char*)d_ws;
    auto carve = [&](size_t bytes) -> void* {
        void* q = (void*)p;
        p += (bytes + 255) & ~(size_t)255;
        return q;
    };
    WS ws;
    ws.xb    = (unsigned short*)carve((size_t)N_ * DT_ * sizeof(unsigned short));
    ws.wpack = (unsigned short*)carve((size_t)3 * DT_ * H_ * sizeof(unsigned short));
    ws.h     = (float*)carve((size_t)N_ * H_ * sizeof(float));
    ws.hb    = (float*)carve((size_t)N_ * H_ * sizeof(float));
    ws.acc   = (float*)carve((size_t)N_ * H_ * sizeof(float));
    ws.cnt   = (float*)carve((size_t)N_ * sizeof(float));
    ws.stats = (float*)carve((size_t)2 * H_ * sizeof(float));
    ws.pacc  = (float*)carve((size_t)B_ * O_ * sizeof(float));
    ws.pcnt  = (float*)carve((size_t)B_ * sizeof(float));

    float* out_t = (float*)d_out;           // [64,128]
    float* out_p = out_t + (size_t)B_ * O_; // [64,128]

    run_branch(x_t, DT_, ei_t, et_t, batch_t,
               w_t1_rel, w_t1_root, b_t1, w_t2_rel, w_t2_root, b_t2,
               g_t1, be_t1, g_t2, be_t2, out_t, ws, stream);

    run_branch(x_p, DP_, ei_p, et_p, batch_p,
               w_p1_rel, w_p1_root, b_p1, w_p2_rel, w_p2_root, b_p2,
               g_p1, be_p1, g_p2, be_p2, out_p, ws, stream);
}
